// SpilloverGNN_21010980012569
// MI455X (gfx1250) — matrix operand order repaired and checked by reading, not compiled, not run on hardware
//
#include <hip/hip_runtime.h>

// ---------------------------------------------------------------------------
// CDNA5 (gfx1250, wave32) GAT forward.
//  - Dense 128x128 GEMMs via v_wmma_f32_16x16x32_bf16, split-bf16 (hi/lo)
//    3-product emulation for ~fp32 accuracy. Precision split is precomputed
//    (O(M*K) elementwise) so the GEMM inner loop is LDS b128 loads + WMMA.
//  - Edge softmax via L2-resident atomics (uint-ordered atomicMax + atomicAdd).
// ---------------------------------------------------------------------------

typedef __attribute__((ext_vector_type(16))) __bf16 v16bf;
typedef __attribute__((ext_vector_type(8)))  __bf16 v8bf;
typedef __attribute__((ext_vector_type(8)))  float  v8f;

#define GAT_N 50000
#define GAT_E 600000
#define GAT_D 128
#define GAT_H 8
#define GAT_L 3

// ---------------------------------------------------------------------------
// Precision-split precompute kernels (hoisted out of the GEMM inner loop).
// ---------------------------------------------------------------------------

// Activations: row-major [M,128] fp32 -> hi/lo bf16, same layout.
__global__ void split_bf16_rows(const float* __restrict__ in,
                                __bf16* __restrict__ hi, __bf16* __restrict__ lo,
                                int n) {
  int i = blockIdx.x * blockDim.x + threadIdx.x;
  if (i >= n) return;
  float f  = in[i];
  __bf16 h = (__bf16)f;
  hi[i] = h;
  lo[i] = (__bf16)(f - (float)h);
}

// Weights: [K=128][N=128] fp32 -> TRANSPOSED hi/lo bf16 ([N][K]) so B-matrix
// WMMA fragments read contiguous K from LDS.
__global__ void split_bf16_T128(const float* __restrict__ in,
                                __bf16* __restrict__ hiT, __bf16* __restrict__ loT) {
  int i = blockIdx.x * blockDim.x + threadIdx.x;  // 0..16383
  if (i >= 128 * 128) return;
  int k = i >> 7, n = i & 127;
  float f  = in[k * 128 + n];
  __bf16 h = (__bf16)f;
  hiT[n * 128 + k] = h;
  loT[n * 128 + k] = (__bf16)(f - (float)h);
}

// ---------------------------------------------------------------------------
// GEMM: C[M,128] = A[M,128] @ B[128,128]  (+bias, +relu per `fuse`)
// Block: 128 threads (4 waves); 64-row x 128-col tile per block;
// each wave owns 16 rows x 8 WMMA accumulators.
// fuse: bit0 = add bias[col], bit1 = relu
// ---------------------------------------------------------------------------
__global__ __launch_bounds__(128) void gemm128_wmma_split(
    const __bf16* __restrict__ Ahi, const __bf16* __restrict__ Alo,
    const __bf16* __restrict__ BhiT, const __bf16* __restrict__ BloT,
    const float* __restrict__ bias, float* __restrict__ C,
    int M, int fuse)
{
  __shared__ __attribute__((aligned(16))) __bf16 lAhi[64][32];   // 4 KB
  __shared__ __attribute__((aligned(16))) __bf16 lAlo[64][32];   // 4 KB
  __shared__ __attribute__((aligned(16))) __bf16 lBhi[128][32];  // 8 KB
  __shared__ __attribute__((aligned(16))) __bf16 lBlo[128][32];  // 8 KB

  const int tid  = threadIdx.x;
  const int wave = tid >> 5;
  const int lane = tid & 31;
  const int row0 = blockIdx.x * 64;

  v8f acc[8];
#pragma unroll
  for (int i = 0; i < 8; ++i) acc[i] = {};

  const int mrow = lane & 15;              // M row (A) / N col (B) within tile
  const int viA  = (lane & 16) ? 2 : 0;    // A K-base 0/16 -> v8bf index 0/2
  const int viB  = (lane & 16) ? 2 : 0;    // B K-base 0/16 -> v8bf index 0/2

  for (int k0 = 0; k0 < 128; k0 += 32) {
    // --- cooperative bf16 tile fills (zero-padded past M) ---
    {
      int r = tid >> 1, cb = (tid & 1) * 16;
      int gr = row0 + r;
      if (gr < M) {
#pragma unroll
        for (int i = 0; i < 16; ++i) {
          lAhi[r][cb + i] = Ahi[(size_t)gr * 128 + k0 + cb + i];
          lAlo[r][cb + i] = Alo[(size_t)gr * 128 + k0 + cb + i];
        }
        // prefetch next K-slab of A (global_prefetch_b8)
        if (k0 + 32 < 128)
          __builtin_prefetch(&Ahi[(size_t)gr * 128 + k0 + 32 + cb], 0, 3);
      } else {
#pragma unroll
        for (int i = 0; i < 16; ++i) {
          lAhi[r][cb + i] = (__bf16)0.0f;
          lAlo[r][cb + i] = (__bf16)0.0f;
        }
      }
#pragma unroll
      for (int i = 0; i < 32; ++i) {       // B: thread t owns column n = t
        lBhi[tid][i] = BhiT[tid * 128 + k0 + i];
        lBlo[tid][i] = BloT[tid * 128 + k0 + i];
      }
    }
    __syncthreads();

    // --- A fragments: two aligned v8bf LDS loads each (ds_load_b128) ---
    const v8bf* pah = (const v8bf*)&lAhi[wave * 16 + mrow][0];
    const v8bf* pal = (const v8bf*)&lAlo[wave * 16 + mrow][0];
    v16bf a_hi = __builtin_shufflevector(pah[viA], pah[viA + 1],
        0, 1, 2, 3, 4, 5, 6, 7, 8, 9, 10, 11, 12, 13, 14, 15);
    v16bf a_lo = __builtin_shufflevector(pal[viA], pal[viA + 1],
        0, 1, 2, 3, 4, 5, 6, 7, 8, 9, 10, 11, 12, 13, 14, 15);

#pragma unroll
    for (int nt = 0; nt < 8; ++nt) {
      const v8bf* pbh = (const v8bf*)&lBhi[nt * 16 + mrow][0];
      const v8bf* pbl = (const v8bf*)&lBlo[nt * 16 + mrow][0];
      v16bf b_hi = __builtin_shufflevector(pbh[viB], pbh[viB + 1],
          0, 1, 2, 3, 4, 5, 6, 7, 8, 9, 10, 11, 12, 13, 14, 15);
      v16bf b_lo = __builtin_shufflevector(pbl[viB], pbl[viB + 1],
          0, 1, 2, 3, 4, 5, 6, 7, 8, 9, 10, 11, 12, 13, 14, 15);
      // D = A*B + C, split-bf16: hi*hi + lo*hi + hi*lo  (~fp32 accuracy)
      acc[nt] = __builtin_amdgcn_wmma_f32_16x16x32_bf16(
          false, a_hi, false, b_hi, (short)0, acc[nt], false, false);
      acc[nt] = __builtin_amdgcn_wmma_f32_16x16x32_bf16(
          false, a_lo, false, b_hi, (short)0, acc[nt], false, false);
      acc[nt] = __builtin_amdgcn_wmma_f32_16x16x32_bf16(
          false, a_hi, false, b_lo, (short)0, acc[nt], false, false);
    }
    __syncthreads();
  }

  // Store: C/D layout = VGPR r -> M=r (lanes 0-15) / M=8+r (lanes 16-31)
  const int rofs = (lane & 16) ? 8 : 0;
#pragma unroll
  for (int nt = 0; nt < 8; ++nt) {
#pragma unroll
    for (int r = 0; r < 8; ++r) {
      int row = row0 + wave * 16 + r + rofs;
      int col = nt * 16 + mrow;
      if (row < M) {
        float v = acc[nt][r];
        if (fuse & 1) v += bias[col];
        if (fuse & 2) v = fmaxf(v, 0.0f);
        C[(size_t)row * 128 + col] = v;
      }
    }
  }
}

// ------------------------- ordered-float encoding for atomicMax ------------
__device__ __forceinline__ unsigned fenc(float f) {
  unsigned u = __float_as_uint(f);
  return (u & 0x80000000u) ? ~u : (u | 0x80000000u);
}
__device__ __forceinline__ float fdec(unsigned u) {
  unsigned b = (u & 0x80000000u) ? (u & 0x7FFFFFFFu) : ~u;
  return __uint_as_float(b);
}
__device__ __forceinline__ float leaky(float f) {
  return f > 0.0f ? f : 0.2f * f;
}

// e_src/e_dst per (node, head); also init segment-max / denom.
__global__ void attn_scores_init(const float* __restrict__ xp,
                                 const float* __restrict__ a_src,
                                 const float* __restrict__ a_dst,
                                 float* __restrict__ e_src,
                                 float* __restrict__ e_dst,
                                 unsigned* __restrict__ m_u,
                                 float* __restrict__ denom, int Nn)
{
  int i = blockIdx.x * blockDim.x + threadIdx.x;
  if (i >= Nn * GAT_H) return;
  int n = i >> 3, hh = i & 7;
  const float* xr = xp + (size_t)n * GAT_D + hh * 16;
  float s = 0.0f, d = 0.0f;
#pragma unroll
  for (int c = 0; c < 16; ++c) {
    float v = xr[c];
    s += v * a_src[hh * 16 + c];
    d += v * a_dst[hh * 16 + c];
  }
  e_src[i] = s;
  e_dst[i] = d;
  m_u[i]   = 0x007FFFFFu;  // fenc(-inf)
  denom[i] = 0.0f;
}

__global__ void zero_f32(float* __restrict__ p, int n) {
  int i = blockIdx.x * blockDim.x + threadIdx.x;
  if (i < n) p[i] = 0.0f;
}

// pass 1: segment max over incoming edges (thread per edge x head)
__global__ void edge_max(const int* __restrict__ srcp, const int* __restrict__ dstp,
                         const float* __restrict__ e_src, const float* __restrict__ e_dst,
                         unsigned* __restrict__ m_u, int E, int Etot)
{
  int i = blockIdx.x * blockDim.x + threadIdx.x;
  if (i >= Etot * GAT_H) return;
  int e = i >> 3, hh = i & 7;
  int s = (e < E) ? srcp[e] : (e - E);
  int d = (e < E) ? dstp[e] : (e - E);
  float ed = leaky(e_src[s * GAT_H + hh] + e_dst[d * GAT_H + hh]);
  atomicMax(&m_u[d * GAT_H + hh], fenc(ed));
}

// pass 2: denom = segment sum of exp(e - m)
__global__ void edge_expsum(const int* __restrict__ srcp, const int* __restrict__ dstp,
                            const float* __restrict__ e_src, const float* __restrict__ e_dst,
                            const unsigned* __restrict__ m_u, float* __restrict__ denom,
                            int E, int Etot)
{
  int i = blockIdx.x * blockDim.x + threadIdx.x;
  if (i >= Etot * GAT_H) return;
  int e = i >> 3, hh = i & 7;
  int s = (e < E) ? srcp[e] : (e - E);
  int d = (e < E) ? dstp[e] : (e - E);
  float ed = leaky(e_src[s * GAT_H + hh] + e_dst[d * GAT_H + hh]);
  float w  = __expf(ed - fdec(m_u[d * GAT_H + hh]));
  atomicAdd(&denom[d * GAT_H + hh], w);
}

// pass 3: h_new[dst, :] += alpha * xp[src, :]  (thread per edge x channel)
__global__ void edge_aggregate(const int* __restrict__ srcp, const int* __restrict__ dstp,
                               const float* __restrict__ e_src, const float* __restrict__ e_dst,
                               const unsigned* __restrict__ m_u, const float* __restrict__ denom,
                               const float* __restrict__ xp, float* __restrict__ hnew,
                               int E, int Etot)
{
  int i = blockIdx.x * blockDim.x + threadIdx.x;
  if (i >= Etot * GAT_D) return;
  int e = i >> 7, dch = i & 127;
  int hh = dch >> 4;
  int s = (e < E) ? srcp[e] : (e - E);
  int d = (e < E) ? dstp[e] : (e - E);
  float ed    = leaky(e_src[s * GAT_H + hh] + e_dst[d * GAT_H + hh]);
  float alpha = __expf(ed - fdec(m_u[d * GAT_H + hh])) / denom[d * GAT_H + hh];
  atomicAdd(&hnew[(size_t)d * GAT_D + dch], alpha * xp[(size_t)s * GAT_D + dch]);
}

// bias + layernorm + relu + residual (block of 128 threads per node, wave32)
__global__ __launch_bounds__(128) void ln_relu_residual(
    const float* __restrict__ hnew, const float* __restrict__ gbias,
    const float* __restrict__ g, const float* __restrict__ b,
    float* __restrict__ h)
{
  __shared__ float red[4];
  int n = blockIdx.x, d = threadIdx.x;
  int wave = d >> 5, lane = d & 31;

  float v = hnew[(size_t)n * GAT_D + d] + gbias[d];

  float s = v;
#pragma unroll
  for (int o = 16; o > 0; o >>= 1) s += __shfl_xor(s, o);
  if (lane == 0) red[wave] = s;
  __syncthreads();
  float mu = (red[0] + red[1] + red[2] + red[3]) * (1.0f / 128.0f);
  __syncthreads();

  float diff = v - mu;
  float q = diff * diff;
#pragma unroll
  for (int o = 16; o > 0; o >>= 1) q += __shfl_xor(q, o);
  if (lane == 0) red[wave] = q;
  __syncthreads();
  float var = (red[0] + red[1] + red[2] + red[3]) * (1.0f / 128.0f);

  float ln = diff * rsqrtf(var + 1e-5f) * g[d] + b[d];
  h[(size_t)n * GAT_D + d] += fmaxf(ln, 0.0f);
}

// out[n] = h[n,:] @ out_W + out_b   (wave per node)
__global__ void out_proj(const float* __restrict__ h, const float* __restrict__ w,
                         const float* __restrict__ b, float* __restrict__ out, int Nn)
{
  int gid  = blockIdx.x * blockDim.x + threadIdx.x;
  int node = gid >> 5, lane = gid & 31;
  if (node >= Nn) return;
  float s = 0.0f;
#pragma unroll
  for (int j = 0; j < 4; ++j)
    s += h[(size_t)node * GAT_D + lane + j * 32] * w[lane + j * 32];
#pragma unroll
  for (int o = 16; o > 0; o >>= 1) s += __shfl_xor(s, o);
  if (lane == 0) out[node] = s + b[0];
}

// ---------------------------------------------------------------------------
extern "C" void kernel_launch(void* const* d_in, const int* in_sizes, int n_in,
                              void* d_out, int out_size, void* d_ws, size_t ws_size,
                              hipStream_t stream) {
  const float* x     = (const float*)d_in[0];
  const int*   ei    = (const int*)d_in[1];   // [2,E]
  const float* in_W  = (const float*)d_in[2];
  const float* in_b  = (const float*)d_in[3];
  const float* gat_W = (const float*)d_in[4]; // [L,128,128]
  const float* a_src = (const float*)d_in[5]; // [L,8,16]
  const float* a_dst = (const float*)d_in[6];
  const float* gat_b = (const float*)d_in[7]; // [L,128]
  const float* ln_g  = (const float*)d_in[8];
  const float* ln_b  = (const float*)d_in[9];
  const float* out_W = (const float*)d_in[10];
  const float* out_b = (const float*)d_in[11];
  float* out = (float*)d_out;

  const int N = GAT_N, E = GAT_E;
  const int Etot = E + N;
  const int* srcp = ei;
  const int* dstp = ei + E;

  // workspace carve-up (~110 MB)
  char* ws = (char*)d_ws;
  size_t off = 0;
  auto carve = [&](size_t bytes) -> void* {
    void* p = ws + off;
    off += (bytes + 255) & ~(size_t)255;
    return p;
  };
  float*    h    = (float*)carve((size_t)N * GAT_D * 4);
  float*    xp   = (float*)carve((size_t)N * GAT_D * 4);
  float*    hnew = (float*)carve((size_t)N * GAT_D * 4);
  __bf16*   aHi  = (__bf16*)carve((size_t)N * GAT_D * 2);
  __bf16*   aLo  = (__bf16*)carve((size_t)N * GAT_D * 2);
  __bf16*   wHiT = (__bf16*)carve((size_t)GAT_D * GAT_D * 2);
  __bf16*   wLoT = (__bf16*)carve((size_t)GAT_D * GAT_D * 2);
  float*    es   = (float*)carve((size_t)N * GAT_H * 4);
  float*    edv  = (float*)carve((size_t)N * GAT_H * 4);
  unsigned* m_u  = (unsigned*)carve((size_t)N * GAT_H * 4);
  float*    den  = (float*)carve((size_t)N * GAT_H * 4);
  (void)ws_size; (void)in_sizes; (void)n_in; (void)out_size;

  const int gemm_blocks  = (N + 63) / 64;
  const int split_blocks = (N * GAT_D + 255) / 256;

  // h = relu(x @ in_W + in_b)
  split_bf16_rows<<<split_blocks, 256, 0, stream>>>(x, aHi, aLo, N * GAT_D);
  split_bf16_T128<<<64, 256, 0, stream>>>(in_W, wHiT, wLoT);
  gemm128_wmma_split<<<gemm_blocks, 128, 0, stream>>>(
      aHi, aLo, wHiT, wLoT, in_b, h, N, 3);

  for (int l = 0; l < GAT_L; ++l) {
    const float* Wl  = gat_W + (size_t)l * GAT_D * GAT_D;
    const float* asl = a_src + (size_t)l * GAT_H * 16;
    const float* adl = a_dst + (size_t)l * GAT_H * 16;
    const float* gbl = gat_b + (size_t)l * GAT_D;
    const float* gl  = ln_g + (size_t)l * GAT_D;
    const float* bl  = ln_b + (size_t)l * GAT_D;

    // xp = h @ W_l  (bias added post-aggregation)
    split_bf16_rows<<<split_blocks, 256, 0, stream>>>(h, aHi, aLo, N * GAT_D);
    split_bf16_T128<<<64, 256, 0, stream>>>(Wl, wHiT, wLoT);
    gemm128_wmma_split<<<gemm_blocks, 128, 0, stream>>>(
        aHi, aLo, wHiT, wLoT, in_b, xp, N, 0);

    attn_scores_init<<<(N * GAT_H + 255) / 256, 256, 0, stream>>>(
        xp, asl, adl, es, edv, m_u, den, N);
    zero_f32<<<(N * GAT_D + 255) / 256, 256, 0, stream>>>(hnew, N * GAT_D);

    edge_max<<<(Etot * GAT_H + 255) / 256, 256, 0, stream>>>(
        srcp, dstp, es, edv, m_u, E, Etot);
    edge_expsum<<<(Etot * GAT_H + 255) / 256, 256, 0, stream>>>(
        srcp, dstp, es, edv, m_u, den, E, Etot);
    edge_aggregate<<<(Etot * GAT_D + 255) / 256, 256, 0, stream>>>(
        srcp, dstp, es, edv, m_u, den, xp, hnew, E, Etot);

    ln_relu_residual<<<N, 128, 0, stream>>>(hnew, gbl, gl, bl, h);
  }

  out_proj<<<(N * 32 + 255) / 256, 256, 0, stream>>>(h, out_W, out_b, out, N);
}
